// SELayer_21053929685330
// MI455X (gfx1250) — matrix operand order repaired and checked by reading, not compiled, not run on hardware
//
#include <hip/hip_runtime.h>
#include <math.h>

typedef __attribute__((ext_vector_type(16))) _Float16 v16h;
typedef __attribute__((ext_vector_type(8)))  float    v8f;
typedef __attribute__((ext_vector_type(4)))  float    f32x4;   // native vec for NT store

#define CCH   128   // channels
#define BSEG  16    // segments
#define HID   8     // hidden dim

// ws layout (floats):
//   [0,    2048)  sums[16][128]
//   [2048, 2064)  counts[16]
//   [2064, 4112)  gate[16][128]
#define WS_SUM   0
#define WS_CNT   2048
#define WS_GATE  2064
#define WS_ZERO_N 2064   // sums + counts must be zeroed each call

// ---------------------------------------------------------------------------
__global__ __launch_bounds__(256)
void zero_ws_kernel(float* __restrict__ ws) {
    int i = blockIdx.x * blockDim.x + threadIdx.x;
    if (i < WS_ZERO_N) ws[i] = 0.0f;
}

// ---------------------------------------------------------------------------
// Pass 1: segment sums + counts. batch_idx is sorted, so each warp walks a
// contiguous row range, accumulating its float4 channel slice in registers
// and flushing to LDS atomics only on segment change (rare).
__global__ __launch_bounds__(256)
void seg_reduce_kernel(const float4* __restrict__ x4,
                       const int*    __restrict__ bidx,
                       float*        __restrict__ gsum,
                       float*        __restrict__ gcnt,
                       int N, int rows_per_warp) {
    __shared__ float sacc[BSEG * CCH];
    __shared__ float scnt[BSEG];

    const int tid = threadIdx.x;
    for (int i = tid; i < BSEG * CCH; i += blockDim.x) sacc[i] = 0.0f;
    if (tid < BSEG) scnt[tid] = 0.0f;
    __syncthreads();

    const int lane = tid & 31;
    const int warp = blockIdx.x * (blockDim.x >> 5) + (tid >> 5);
    const int r0 = warp * rows_per_warp;
    int r1 = r0 + rows_per_warp;
    if (r1 > N) r1 = N;

    int   cur = -1;
    float ax = 0.f, ay = 0.f, az = 0.f, aw = 0.f, cnt = 0.f;

    for (int r = r0; r < r1; ++r) {
        const int seg = bidx[r];                 // uniform across the warp
        if (seg != cur) {
            if (cur >= 0) {
                float* p = &sacc[cur * CCH + lane * 4];
                atomicAdd(p + 0, ax); atomicAdd(p + 1, ay);
                atomicAdd(p + 2, az); atomicAdd(p + 3, aw);
                if (lane == 0) atomicAdd(&scnt[cur], cnt);
            }
            cur = seg; ax = ay = az = aw = 0.f; cnt = 0.f;
        }
        const float4 v = x4[(size_t)r * 32 + lane];   // 512B/row, b128 loads
        ax += v.x; ay += v.y; az += v.z; aw += v.w;
        cnt += 1.0f;
    }
    if (cur >= 0) {
        float* p = &sacc[cur * CCH + lane * 4];
        atomicAdd(p + 0, ax); atomicAdd(p + 1, ay);
        atomicAdd(p + 2, az); atomicAdd(p + 3, aw);
        if (lane == 0) atomicAdd(&scnt[cur], cnt);
    }
    __syncthreads();

    for (int i = tid; i < BSEG * CCH; i += blockDim.x) {
        const float v = sacc[i];
        if (v != 0.0f) atomicAdd(&gsum[i], v);
    }
    if (tid < BSEG) {
        const float v = scnt[tid];
        if (v != 0.0f) atomicAdd(&gcnt[tid], v);
    }
}

// ---------------------------------------------------------------------------
// Pass 2: one wave does the whole SE MLP with v_wmma_f32_16x16x32_f16.
// All operands are staged unconditionally into zero-padded f16 LDS arrays so
// fragment packing is pure branch-free ds_load (no predicated global loads).
//
// A-fragment (16x32 f16): lane L holds row M=L%16; half h maps to
//   K = h + (L>=16 ? 16 : 0)
// B-fragment (32x16 f16): lane L holds col N=L%16; half h maps to
//   K = h + (L>=16 ? 16 : 0)
// C/D (16x16 f32): lane L holds N=L%16; element v holds M = v + (L>=16 ? 8 : 0)
__global__ __launch_bounds__(32)
void se_mlp_kernel(const float* __restrict__ gsum,
                   const float* __restrict__ gcnt,
                   const float* __restrict__ W1,  // [128,8] row-major
                   const float* __restrict__ b1,  // [8]
                   const float* __restrict__ W2,  // [8,128] row-major
                   const float* __restrict__ b2,  // [128]
                   float* __restrict__ gate) {    // [16,128]
    __shared__ _Float16 sA [BSEG * CCH];   // mean, f16            (4 KB)
    __shared__ _Float16 sB1[CCH * 16];     // W1 padded N 8->16    (4 KB)
    __shared__ _Float16 sA2[BSEG * 32];    // h padded K 8->32     (1 KB)
    __shared__ _Float16 sB2[32 * CCH];     // W2 padded K 8->32    (8 KB)
    __shared__ float    scc[BSEG];         // counts

    const int lane   = threadIdx.x;
    const int M      = lane & 15;
    const int hiHalf = lane >> 4;   // 0: lanes 0-15, 1: lanes 16-31

    // ---- zero the padded staging arrays ----
    for (int i = lane; i < CCH * 16;  i += 32) sB1[i] = (_Float16)0.0f;
    for (int i = lane; i < BSEG * 32; i += 32) sA2[i] = (_Float16)0.0f;
    for (int i = lane; i < 32 * CCH;  i += 32) sB2[i] = (_Float16)0.0f;
    if (lane < BSEG) scc[lane] = fmaxf(gcnt[lane], 1.0f);
    __syncthreads();

    // ---- stage mean (f16) and weights (f16, padded) cooperatively ----
    for (int i = lane; i < BSEG * CCH; i += 32) {
        const int seg = i >> 7;
        sA[i] = (_Float16)(gsum[i] / scc[seg]);
    }
    for (int i = lane; i < CCH * HID; i += 32) {   // W1: [c][h] -> sB1[c][h]
        const int c = i >> 3, n = i & 7;
        sB1[c * 16 + n] = (_Float16)W1[i];
    }
    for (int i = lane; i < HID * CCH; i += 32) {   // W2: [k][ch] -> sB2[k][ch]
        sB2[i] = (_Float16)W2[i];                  // k<8 rows; rows 8..31 stay 0
    }
    __syncthreads();

    // ---- mean[16,128] @ W1[128,8->16], K split into 4x32 ----
    v8f c1 = {};
    #pragma unroll
    for (int chunk = 0; chunk < 4; ++chunk) {
        v16h a, b;
        #pragma unroll
        for (int h = 0; h < 16; ++h) {
            const int ka = h + (hiHalf ? 16 : 0);
            a[h] = sA[M * CCH + chunk * 32 + ka];
            const int kb = h + (hiHalf ? 16 : 0);
            b[h] = sB1[(chunk * 32 + kb) * 16 + M];
        }
        c1 = __builtin_amdgcn_wmma_f32_16x16x32_f16(false, a, false, b,
                                                    (short)0, c1, false, false);
    }

    // bias + ReLU, stage h[16][8] into zero-padded sA2 (cross-lane transpose)
    if (M < HID) {
        const float bb = b1[M];
        #pragma unroll
        for (int v = 0; v < 8; ++v) {
            const int m = v + (hiHalf ? 8 : 0);
            sA2[m * 32 + M] = (_Float16)fmaxf(c1[v] + bb, 0.0f);
        }
    }
    __syncthreads();

    // ---- h[16,K=32 padded] @ W2[8->32,128], 8 N-tiles of 16 ----
    v16h a2;
    #pragma unroll
    for (int h = 0; h < 16; ++h) {
        const int ka = h + (hiHalf ? 16 : 0);
        a2[h] = sA2[M * 32 + ka];
    }
    #pragma unroll
    for (int t = 0; t < 8; ++t) {
        const int ch = t * 16 + M;
        v16h bf;
        #pragma unroll
        for (int h = 0; h < 16; ++h) {
            const int kb = h + (hiHalf ? 16 : 0);
            bf[h] = sB2[kb * CCH + ch];
        }
        v8f c2 = {};
        c2 = __builtin_amdgcn_wmma_f32_16x16x32_f16(false, a2, false, bf,
                                                    (short)0, c2, false, false);
        const float bias = b2[ch];
        #pragma unroll
        for (int v = 0; v < 8; ++v) {
            const int m = v + (hiHalf ? 8 : 0);
            const float z = c2[v] + bias;
            gate[m * CCH + ch] = 1.0f / (1.0f + __expf(-z));
        }
    }
}

// ---------------------------------------------------------------------------
// Pass 3: out = x * gate[batch_idx], pure float4 stream (HBM-bound).
__global__ __launch_bounds__(256)
void apply_gate_kernel(const float4* __restrict__ x4,
                       const int*    __restrict__ bidx,
                       const float4* __restrict__ gate4,  // [16*32]
                       float4*       __restrict__ out4,
                       long long total4) {
    const long long stride = (long long)gridDim.x * blockDim.x;
    for (long long g = (long long)blockIdx.x * blockDim.x + threadIdx.x;
         g < total4; g += stride) {
        const long long r   = g >> 5;         // row (32 float4 per row)
        const int       c4  = (int)(g & 31);  // float4 index within row
        const int       seg = bidx[r];
        const float4 gv = gate4[seg * 32 + c4];   // 8KB table, stays in cache
        float4 v = x4[g];
        f32x4 nv;
        nv.x = v.x * gv.x;
        nv.y = v.y * gv.y;
        nv.z = v.z * gv.z;
        nv.w = v.w * gv.w;
        // output is never re-read -> non-temporal b128 store
        __builtin_nontemporal_store(nv, (f32x4*)&out4[g]);
    }
}

// ---------------------------------------------------------------------------
extern "C" void kernel_launch(void* const* d_in, const int* in_sizes, int n_in,
                              void* d_out, int out_size, void* d_ws, size_t ws_size,
                              hipStream_t stream) {
    const float* x    = (const float*)d_in[0];
    const int*   bidx = (const int*)  d_in[1];
    const float* W1   = (const float*)d_in[2];
    const float* b1   = (const float*)d_in[3];
    const float* W2   = (const float*)d_in[4];
    const float* b2   = (const float*)d_in[5];
    float*       out  = (float*)d_out;
    float*       ws   = (float*)d_ws;

    const int N = in_sizes[0] / CCH;

    float* gsum = ws + WS_SUM;
    float* gcnt = ws + WS_CNT;
    float* gate = ws + WS_GATE;

    // 0) zero accumulators (deterministic per call)
    zero_ws_kernel<<<(WS_ZERO_N + 255) / 256, 256, 0, stream>>>(ws);

    // 1) segment reduce
    const int blocks = 1024;                    // 8 waves/block
    const int warps  = blocks * 8;
    const int rpw    = (N + warps - 1) / warps;
    seg_reduce_kernel<<<blocks, 256, 0, stream>>>(
        (const float4*)x, bidx, gsum, gcnt, N, rpw);

    // 2) SE MLP on one wave (WMMA)
    se_mlp_kernel<<<1, 32, 0, stream>>>(gsum, gcnt, W1, b1, W2, b2, gate);

    // 3) broadcast multiply
    const long long total4 = (long long)N * (CCH / 4);
    apply_gate_kernel<<<4096, 256, 0, stream>>>(
        (const float4*)x, bidx, (const float4*)gate, (float4*)out, total4);
}
